// PredicteurStockQuotidien_28269474742919
// MI455X (gfx1250) — matrix operand order repaired and checked by reading, not compiled, not run on hardware
//
#include <hip/hip_runtime.h>

// Problem constants (match reference)
#define T_LEN 4096
#define B_SZ  2048
#define H_DIM 10

typedef __attribute__((ext_vector_type(16))) _Float16     v16h;
typedef __attribute__((ext_vector_type(8)))  float        v8f;
typedef __attribute__((ext_vector_type(8)))  unsigned int v8u;

#if __has_builtin(__builtin_amdgcn_tanhf)
__device__ __forceinline__ float fast_tanh(float x) { return __builtin_amdgcn_tanhf(x); }
#else
__device__ __forceinline__ float fast_tanh(float x) { return tanhf(x); }
#endif

__device__ __forceinline__ float sigm(float x) {
  // sigmoid(x) = 0.5*tanh(0.5x)+0.5
  return fmaf(0.5f, fast_tanh(0.5f * x), 0.5f);
}

__device__ __forceinline__ unsigned pk(float a, float b) {
  // v_cvt_pk_rtz_f16_f32 -> packed f16x2 as a dword
  return __builtin_bit_cast(unsigned, __builtin_amdgcn_cvt_pkrtz(a, b));
}

__device__ __forceinline__ unsigned swz16(unsigned v) {
  // ds_swizzle_b32 SWAPX16: xor_mask=0x10, and_mask=0x1f -> exchange half-waves
  return (unsigned)__builtin_amdgcn_ds_swizzle((int)v, 0x401F);
}

// One wave32 per 16 batch elements. Weights live in registers in WMMA A-layout.
// Gate-row permutation: tile tau, row r (0..15): gate = r&3,
//   h = (r<8 ? 0 : 6) + 2*tau + ((r&7)>>2)     (h>=10 rows are zero padding)
// => after WMMA, lane n<16 holds i/f/g/o for h in {0..5} of batch n,
//    lane n+16 holds them for h in {6..9}. Activations are lane-local.
__global__ __launch_bounds__(32)
void PredicteurStockQuotidien_lstm_wmma_kernel(
    const float* __restrict__ x,
    const float* __restrict__ Wih0, const float* __restrict__ Whh0,
    const float* __restrict__ bih0, const float* __restrict__ bhh0,
    const float* __restrict__ Wih1, const float* __restrict__ Whh1,
    const float* __restrict__ bih1, const float* __restrict__ bhh1,
    const float* __restrict__ Wlin, const float* __restrict__ blin,
    float* __restrict__ out)
{
  const int  lane = (int)(threadIdx.x & 31);
  const int  m    = lane & 15;
  const bool hi   = lane >= 16;
  const int  b    = (int)blockIdx.x * 16 + m;

  // ---- Build constant A tiles (f16, WMMA 16x32 A-layout).
  // Layer 0 K layout: K<10 -> W_hh0, K==10 -> W_ih0 (x column), K==11 -> bias sum.
  // Layer 1 K layout: K<10 -> W_ih1 (h0 input), 10<=K<20 -> W_hh1, K==20 -> bias sum.
  v16h A0[3], A1[3];
  {
    const int  r    = m;
    const int  gate = r & 3;
#pragma unroll
    for (int tau = 0; tau < 3; ++tau) {
      const int  h   = ((r < 8) ? 0 : 6) + 2 * tau + ((r & 7) >> 2);
      const bool ok  = (h < H_DIM);
      const int  row = gate * H_DIM + h;
#pragma unroll
      for (int j = 0; j < 16; ++j) {
        const int K = j + (hi ? 16 : 0);
        float v0 = 0.f, v1 = 0.f;
        if (ok) {
          if (K < 10)       v0 = Whh0[row * H_DIM + K];
          else if (K == 10) v0 = Wih0[row];                 // IN_DIM == 1
          else if (K == 11) v0 = bih0[row] + bhh0[row];
          if (K < 10)       v1 = Wih1[row * H_DIM + K];
          else if (K < 20)  v1 = Whh1[row * H_DIM + (K - 10)];
          else if (K == 20) v1 = bih1[row] + bhh1[row];
        }
        A0[tau][j] = (_Float16)v0;
        A1[tau][j] = (_Float16)v1;
      }
    }
  }

  // Final-projection coefficients, per lane (lane-lo: h 0..5, lane-hi: h 6..9).
  float wl[6];
#pragma unroll
  for (int l = 0; l < 6; ++l) {
    const int h = (hi ? 6 : 0) + l;
    wl[l] = (h < H_DIM) ? Wlin[h] : 0.f;
  }
  const float bl = blin[0];

  // ---- Recurrent state (all registers; c-state never crosses lanes).
  unsigned hp0[5] = {0u, 0u, 0u, 0u, 0u};  // lane-lo: packed f16 h0[0..9]
  float c0[6]  = {0.f, 0.f, 0.f, 0.f, 0.f, 0.f};
  float h1f[6] = {0.f, 0.f, 0.f, 0.f, 0.f, 0.f};
  float c1[6]  = {0.f, 0.f, 0.f, 0.f, 0.f, 0.f};

  const float*   xp    = x + b;        // x[t*B + b]
  float          xcur  = xp[0];
  const unsigned ONE_H = 0x3C00u;      // f16 1.0 in low half

  for (int t = 0; t < T_LEN; ++t) {
    // Prefetch next step's x (register double-buffer hides load latency).
    const int   tn    = (t + 1 < T_LEN) ? (t + 1) : t;
    const float xnext = xp[(size_t)tn * B_SZ];

    // ---- Layer 0: gates0 = [W_hh0 | W_ih0 | bias] x [h0 ; x ; 1]  (one WMMA per tile)
    v8u b0u;
    b0u[0] = hi ? 0u : hp0[0];
    b0u[1] = hi ? 0u : hp0[1];
    b0u[2] = hi ? 0u : hp0[2];
    b0u[3] = hi ? 0u : hp0[3];
    b0u[4] = hi ? 0u : hp0[4];
    b0u[5] = hi ? 0u : pk(xcur, 1.0f);   // K=10: x, K=11: 1.0
    b0u[6] = 0u;
    b0u[7] = 0u;
    const v16h B0 = __builtin_bit_cast(v16h, b0u);

    v8f g0[3];
#pragma unroll
    for (int tau = 0; tau < 3; ++tau) {
      v8f cz = {};
      g0[tau] = __builtin_amdgcn_wmma_f32_16x16x32_f16(
          false, A0[tau], false, B0, (short)0, cz, false, false);
    }

    float hn[6];
#pragma unroll
    for (int l = 0; l < 6; ++l) {
      const int tau = l >> 1, s = (l & 1) * 4;
      const float gi = sigm(g0[tau][s + 0]);
      const float gf = sigm(g0[tau][s + 1]);
      const float gg = fast_tanh(g0[tau][s + 2]);
      const float go = sigm(g0[tau][s + 3]);
      const float c  = fmaf(gf, c0[l], gi * gg);
      c0[l] = c;
      hn[l] = go * fast_tanh(c);
    }

    // Repack h0: lane-lo gathers h0[6..9] from the upper half-wave (2 swizzles).
    {
      const unsigned p01 = pk(hn[0], hn[1]);
      const unsigned p23 = pk(hn[2], hn[3]);
      const unsigned p45 = pk(hn[4], hn[5]);
      hp0[0] = p01;
      hp0[1] = p23;
      hp0[2] = p45;
      hp0[3] = swz16(p01);   // lane-lo receives hi-half's (h6,h7)
      hp0[4] = swz16(p23);   // lane-lo receives hi-half's (h8,h9)
    }

    // ---- Layer 1: gates1 = [W_ih1 | W_hh1 | bias] x [h0_new ; h1 ; 1]
    const unsigned q01 = pk(h1f[0], h1f[1]);
    const unsigned q23 = pk(h1f[2], h1f[3]);
    const unsigned q45 = pk(h1f[4], h1f[5]);
    v8u b1u;
    b1u[0] = hi ? q01   : hp0[0];   // hi: K16,17 = h1[6],h1[7]
    b1u[1] = hi ? q23   : hp0[1];   // hi: K18,19 = h1[8],h1[9]
    b1u[2] = hi ? ONE_H : hp0[2];   // hi: K20 = 1.0 (bias column)
    b1u[3] = hi ? 0u    : hp0[3];
    b1u[4] = hi ? 0u    : hp0[4];
    b1u[5] = hi ? 0u    : q01;      // lo: K10,11 = h1[0],h1[1]
    b1u[6] = hi ? 0u    : q23;
    b1u[7] = hi ? 0u    : q45;
    const v16h B1 = __builtin_bit_cast(v16h, b1u);

    v8f g1[3];
#pragma unroll
    for (int tau = 0; tau < 3; ++tau) {
      v8f cz = {};
      g1[tau] = __builtin_amdgcn_wmma_f32_16x16x32_f16(
          false, A1[tau], false, B1, (short)0, cz, false, false);
    }

#pragma unroll
    for (int l = 0; l < 6; ++l) {
      const int tau = l >> 1, s = (l & 1) * 4;
      const float gi = sigm(g1[tau][s + 0]);
      const float gf = sigm(g1[tau][s + 1]);
      const float gg = fast_tanh(g1[tau][s + 2]);
      const float go = sigm(g1[tau][s + 3]);
      const float c  = fmaf(gf, c1[l], gi * gg);
      c1[l] = c;
      h1f[l] = go * fast_tanh(c);
    }

    // ---- Final projection only for the last two timesteps
    //      (reshape(B,T,H)[-1] == concat(h2[T-2], h2[T-1]))
    if (t >= T_LEN - 2) {
      float acc = 0.f;
#pragma unroll
      for (int l = 0; l < 6; ++l) acc = fmaf(h1f[l], wl[l], acc);
      const float other =
          __builtin_bit_cast(float, swz16(__builtin_bit_cast(unsigned, acc)));
      if (!hi) {
        const int row = (t - (T_LEN - 2)) * B_SZ + b;
        out[row] = acc + other + bl;
      }
    }

    xcur = xnext;
  }
}

extern "C" void kernel_launch(void* const* d_in, const int* in_sizes, int n_in,
                              void* d_out, int out_size, void* d_ws, size_t ws_size,
                              hipStream_t stream) {
  const float* x    = (const float*)d_in[0];
  const float* Wih0 = (const float*)d_in[1];
  const float* Whh0 = (const float*)d_in[2];
  const float* bih0 = (const float*)d_in[3];
  const float* bhh0 = (const float*)d_in[4];
  const float* Wih1 = (const float*)d_in[5];
  const float* Whh1 = (const float*)d_in[6];
  const float* bih1 = (const float*)d_in[7];
  const float* bhh1 = (const float*)d_in[8];
  const float* Wlin = (const float*)d_in[9];
  const float* blin = (const float*)d_in[10];
  float* out = (float*)d_out;

  dim3 grid(B_SZ / 16);  // 128 independent wave32's
  dim3 block(32);
  hipLaunchKernelGGL(PredicteurStockQuotidien_lstm_wmma_kernel, grid, block, 0, stream,
                     x, Wih0, Whh0, bih0, bhh0, Wih1, Whh1, bih1, bhh1, Wlin, blin, out);
}
